// MultiHeadAttention_60172491816987
// MI455X (gfx1250) — matrix operand order repaired and checked by reading, not compiled, not run on hardware
//
#include <hip/hip_runtime.h>
#include <hip/hip_bf16.h>

#define D_MODEL 1024
#define NHEAD   16
#define DK      64
#define DV      64
#define BATCH   2
#define SEQ     2048

typedef __bf16 bf16;
typedef __bf16 v16bf __attribute__((ext_vector_type(16)));
typedef __bf16 v8bf  __attribute__((ext_vector_type(8)));
typedef float  v8f   __attribute__((ext_vector_type(8)));

static __device__ __forceinline__ v8f wmma_bf16(v16bf a, v16bf b, v8f c) {
    return __builtin_amdgcn_wmma_f32_16x16x32_bf16(false, a, false, b, (short)0, c,
                                                   false, false);
}

// ds_swizzle xor-mask shuffle (group-of-32 mode, and_mask=0x1f)
template <int IMM>
static __device__ __forceinline__ float swz(float x) {
    int i = __builtin_amdgcn_ds_swizzle(__builtin_bit_cast(int, x), IMM);
    return __builtin_bit_cast(float, i);
}

// A fragment (16x32 bf16) from an f32 row-major row pointer (already offset to k0).
// Lane holds row r = lane%16; elements e=0..15 -> K=16g+e.
static __device__ __forceinline__ v16bf load_a_f32(const float* pr, int g) {
    v16bf a;
#pragma unroll
    for (int e = 0; e < 8; ++e) {
        a[e]     = (bf16)pr[16 * g + e];
        a[e + 8] = (bf16)pr[16 * g + 8 + e];
    }
    return a;
}

// Same fragment from a bf16 row-major row pointer (generic: global or LDS).
static __device__ __forceinline__ v16bf load_a_bf16(const bf16* pr, int g) {
    v8bf lo = *(const v8bf*)(pr + 16 * g);
    v8bf hi = *(const v8bf*)(pr + 16 * g + 8);
    v16bf a;
#pragma unroll
    for (int e = 0; e < 8; ++e) {
        a[e]     = lo[e];
        a[e + 8] = hi[e];
    }
    return a;
}

// Same, but multiply each element by a power-of-two scale (exact in bf16).
static __device__ __forceinline__ v16bf load_a_bf16_scaled(const bf16* pr, int g,
                                                           float s) {
    v8bf lo = *(const v8bf*)(pr + 16 * g);
    v8bf hi = *(const v8bf*)(pr + 16 * g + 8);
    v16bf a;
#pragma unroll
    for (int e = 0; e < 8; ++e) {
        a[e]     = (bf16)((float)lo[e] * s);
        a[e + 8] = (bf16)((float)hi[e] * s);
    }
    return a;
}

// ---------------------------------------------------------------------------
// Kernel 1: f32 [H][K][N] -> bf16 [H][N][K]  (weight transpose + downconvert)
// ---------------------------------------------------------------------------
__global__ void transpose_to_bf16(const float* __restrict__ W, bf16* __restrict__ WT,
                                  int H, int K, int Nc) {
    long total = (long)H * K * Nc;
    for (long t = (long)blockIdx.x * blockDim.x + threadIdx.x; t < total;
         t += (long)gridDim.x * blockDim.x) {
        int  k = (int)(t % K);
        long r = t / K;
        int  n = (int)(r % Nc);
        int  h = (int)(r / Nc);
        WT[t] = (bf16)W[((long)h * K + k) * Nc + n];
    }
}

// ---------------------------------------------------------------------------
// Kernel 2: per-head projection  X[b] (SEQ x 1024) * W[h] (1024 x 64) + bias
// One wave -> 16x64 output tile.  TRANSPOSE_OUT: 0 -> [B][H][SEQ][64] bf16
//                                               1 -> [B][H][64][SEQ] bf16
// ---------------------------------------------------------------------------
template <int TRANSPOSE_OUT>
__global__ void proj_kernel(const float* __restrict__ X,    // [B][SEQ][D_MODEL]
                            const bf16* __restrict__ WT,    // [H][64][D_MODEL]
                            const float* __restrict__ bias, // [H][64]
                            bf16* __restrict__ out) {
    const int lane = threadIdx.x & 31;
    const int g    = lane >> 4;
    const int n    = lane & 15;
    const int wave = blockIdx.x * (blockDim.x >> 5) + (threadIdx.x >> 5);

    const int rowblocks = SEQ / 16;
    const int rb   = wave % rowblocks;
    const int bh   = wave / rowblocks;
    const int h    = bh % NHEAD;
    const int b    = bh / NHEAD;
    const int row0 = rb * 16;

    const float* Xrow = X + ((long)b * SEQ + row0 + n) * D_MODEL; // A row r = n
    const bf16*  Wh   = WT + (long)h * 64 * D_MODEL;

    v8f acc[4] = {v8f{}, v8f{}, v8f{}, v8f{}};
#pragma unroll 4
    for (int k0 = 0; k0 < D_MODEL; k0 += 32) {
        v16bf a = load_a_f32(Xrow + k0, g);
#pragma unroll
        for (int f = 0; f < 4; ++f) {
            v16bf bfrag = *(const v16bf*)(Wh + (long)(16 * f + n) * D_MODEL + k0 + 16 * g);
            acc[f] = wmma_bf16(a, bfrag, acc[f]);
        }
    }

#pragma unroll
    for (int f = 0; f < 4; ++f) {
        float bv = bias[h * 64 + 16 * f + n];
        if (TRANSPOSE_OUT) {
            v8bf pack;
#pragma unroll
            for (int j = 0; j < 8; ++j) pack[j] = (bf16)(acc[f][j] + bv);
            bf16* dst = out + ((long)(b * NHEAD + h) * 64 + 16 * f + n) * SEQ
                            + row0 + 8 * g;                  // 8 contiguous rows
            *(v8bf*)dst = pack;
        } else {
#pragma unroll
            for (int j = 0; j < 8; ++j) {
                int r = row0 + 8 * g + j;
                out[((long)(b * NHEAD + h) * SEQ + r) * 64 + 16 * f + n] =
                    (bf16)(acc[f][j] + bv);
            }
        }
    }
}

// ---------------------------------------------------------------------------
// Kernel 3: flash attention.  One wave owns 16 query rows of one (b,h),
// streams keys/values in blocks of 64 with online softmax (base-2 domain).
// Row-sum l computed on the matrix pipe via a P x ones WMMA tile.
// ---------------------------------------------------------------------------
__global__ void attn_kernel(const bf16* __restrict__ qb,   // [B][H][SEQ][64]
                            const bf16* __restrict__ kbuf, // [B][H][SEQ][64]
                            const bf16* __restrict__ vT,   // [B][H][64][SEQ]
                            bf16* __restrict__ concat) {   // [B][SEQ][H*64]
    __shared__ __align__(16) bf16 ptile[8][16 * 64]; // per-wave 16x64 P tile

    const int lane  = threadIdx.x & 31;
    const int g     = lane >> 4;
    const int n     = lane & 15;
    const int wslot = threadIdx.x >> 5;
    const int wave  = blockIdx.x * (blockDim.x >> 5) + wslot;

    const int rowblocks = SEQ / 16;
    const int rb   = wave % rowblocks;
    const int bh   = wave / rowblocks;
    const int h    = bh % NHEAD;
    const int b    = bh / NHEAD;
    const int row0 = rb * 16;

    const bf16* qbh = qb   + (long)bh * SEQ * 64;
    const bf16* kbh = kbuf + (long)bh * SEQ * 64;
    const bf16* vbh = vT   + (long)bh * 64 * SEQ;

    // Q A-fragments pre-scaled by 1/sqrt(dk)=0.125 (power of two: exact in bf16)
    const bf16* qrow = qbh + (long)(row0 + n) * 64;
    v16bf qa0 = load_a_bf16_scaled(qrow, g, 0.125f);
    v16bf qa1 = load_a_bf16_scaled(qrow + 32, g, 0.125f);

    // ones B-fragment for the row-sum tile
    v16bf ones;
#pragma unroll
    for (int e = 0; e < 16; ++e) ones[e] = (bf16)1.0f;

    v8f o0 = {}, o1 = {}, o2 = {}, o3 = {}, lacc = {};
    float m2[8]; // running max in log2 domain
#pragma unroll
    for (int j = 0; j < 8; ++j) m2[j] = -INFINITY;

    const float LOG2E = 1.44269504088896f;
    bf16* myp = &ptile[wslot][0];

    for (int kb0 = 0; kb0 < SEQ; kb0 += 64) {
        // ---- scores: four 16-key sub-tiles, each = 2 WMMAs over dk ----
        v8f s[4];
#pragma unroll
        for (int s16 = 0; s16 < 4; ++s16) {
            const bf16* krow = kbh + (long)(kb0 + 16 * s16 + n) * 64;
            v16bf kf0 = *(const v16bf*)(krow + 16 * g);
            v16bf kf1 = *(const v16bf*)(krow + 32 + 16 * g);
            v8f sc = {};
            sc = wmma_bf16(qa0, kf0, sc);
            sc = wmma_bf16(qa1, kf1, sc);
            s[s16] = sc;
        }

        // ---- to log2 domain + local row max over 64 keys ----
        float rm[8], tr[8];
#pragma unroll
        for (int x = 0; x < 4; ++x)
#pragma unroll
            for (int j = 0; j < 8; ++j) s[x][j] *= LOG2E;
#pragma unroll
        for (int j = 0; j < 8; ++j)
            rm[j] = fmaxf(fmaxf(s[0][j], s[1][j]), fmaxf(s[2][j], s[3][j]));

        // batched xor-swizzle max reduction across the 16 lanes of each row
#pragma unroll
        for (int j = 0; j < 8; ++j) tr[j] = swz<0x041f>(rm[j]);
#pragma unroll
        for (int j = 0; j < 8; ++j) rm[j] = fmaxf(rm[j], tr[j]);
#pragma unroll
        for (int j = 0; j < 8; ++j) tr[j] = swz<0x081f>(rm[j]);
#pragma unroll
        for (int j = 0; j < 8; ++j) rm[j] = fmaxf(rm[j], tr[j]);
#pragma unroll
        for (int j = 0; j < 8; ++j) tr[j] = swz<0x101f>(rm[j]);
#pragma unroll
        for (int j = 0; j < 8; ++j) rm[j] = fmaxf(rm[j], tr[j]);
#pragma unroll
        for (int j = 0; j < 8; ++j) tr[j] = swz<0x201f>(rm[j]);
#pragma unroll
        for (int j = 0; j < 8; ++j) rm[j] = fmaxf(rm[j], tr[j]);

        // ---- online rescale + P = exp2(s - m), stashed to LDS ----
#pragma unroll
        for (int j = 0; j < 8; ++j) {
            float mn   = fmaxf(m2[j], rm[j]);
            float corr = __builtin_amdgcn_exp2f(m2[j] - mn);
            m2[j] = mn;
            o0[j] *= corr; o1[j] *= corr; o2[j] *= corr; o3[j] *= corr;
            lacc[j] *= corr;
            bf16* prow = myp + (8 * g + j) * 64;
#pragma unroll
            for (int x = 0; x < 4; ++x) {
                float p = __builtin_amdgcn_exp2f(s[x][j] - mn);
                prow[16 * x + n] = (bf16)p;
            }
        }

        // same-wave LDS store->load: DS ops are in-order; wait to be safe
        asm volatile("s_wait_dscnt 0x0" ::: "memory");
        __builtin_amdgcn_wave_barrier();

        // ---- P (16x64) x V (64x64) = 8 WMMAs, + 2 for the row-sum tile ----
        v16bf pa0 = load_a_bf16(myp + n * 64, g);       // keys kb0 .. kb0+31
        v16bf pa1 = load_a_bf16(myp + n * 64 + 32, g);  // keys kb0+32 .. kb0+63
        lacc = wmma_bf16(pa0, ones, lacc);
        lacc = wmma_bf16(pa1, ones, lacc);
        o0 = wmma_bf16(pa0, *(const v16bf*)(vbh + (long)(0  + n) * SEQ + kb0 + 16 * g), o0);
        o1 = wmma_bf16(pa0, *(const v16bf*)(vbh + (long)(16 + n) * SEQ + kb0 + 16 * g), o1);
        o2 = wmma_bf16(pa0, *(const v16bf*)(vbh + (long)(32 + n) * SEQ + kb0 + 16 * g), o2);
        o3 = wmma_bf16(pa0, *(const v16bf*)(vbh + (long)(48 + n) * SEQ + kb0 + 16 * g), o3);
        o0 = wmma_bf16(pa1, *(const v16bf*)(vbh + (long)(0  + n) * SEQ + kb0 + 32 + 16 * g), o0);
        o1 = wmma_bf16(pa1, *(const v16bf*)(vbh + (long)(16 + n) * SEQ + kb0 + 32 + 16 * g), o1);
        o2 = wmma_bf16(pa1, *(const v16bf*)(vbh + (long)(32 + n) * SEQ + kb0 + 32 + 16 * g), o2);
        o3 = wmma_bf16(pa1, *(const v16bf*)(vbh + (long)(48 + n) * SEQ + kb0 + 32 + 16 * g), o3);
    }

    // ---- epilogue: normalize and write concat[b][row][h*64 + col] ----
#pragma unroll
    for (int j = 0; j < 8; ++j) {
        float inv = 1.0f / lacc[j];
        int   r   = row0 + 8 * g + j;
        bf16* crow = concat + ((long)b * SEQ + r) * (NHEAD * DV) + h * 64;
        crow[0  + n] = (bf16)(o0[j] * inv);
        crow[16 + n] = (bf16)(o1[j] * inv);
        crow[32 + n] = (bf16)(o2[j] * inv);
        crow[48 + n] = (bf16)(o3[j] * inv);
    }
}

// ---------------------------------------------------------------------------
// Kernel 4: output projection  concat (B*SEQ x 1024) * Wo (1024 x 1024) + bo
// One wave -> 16x64 f32 output tile.
// ---------------------------------------------------------------------------
__global__ void outproj_kernel(const bf16* __restrict__ concat, // [B*SEQ][1024]
                               const bf16* __restrict__ WoT,    // [1024][1024]
                               const float* __restrict__ bo,    // [1024]
                               float* __restrict__ out) {       // [B*SEQ][1024]
    const int lane = threadIdx.x & 31;
    const int g    = lane >> 4;
    const int n    = lane & 15;
    const int wave = blockIdx.x * (blockDim.x >> 5) + (threadIdx.x >> 5);

    const int colTiles = D_MODEL / 64; // 16
    const int ct   = wave % colTiles;
    const int rb   = wave / colTiles;
    const int row0 = rb * 16;
    const int col0 = ct * 64;

    const bf16* arow = concat + (long)(row0 + n) * D_MODEL;

    v8f acc[4] = {v8f{}, v8f{}, v8f{}, v8f{}};
#pragma unroll 4
    for (int k0 = 0; k0 < D_MODEL; k0 += 32) {
        v16bf a = load_a_bf16(arow + k0, g);
#pragma unroll
        for (int f = 0; f < 4; ++f) {
            v16bf bfrag =
                *(const v16bf*)(WoT + (long)(col0 + 16 * f + n) * D_MODEL + k0 + 16 * g);
            acc[f] = wmma_bf16(a, bfrag, acc[f]);
        }
    }

#pragma unroll
    for (int f = 0; f < 4; ++f) {
        float bv = bo[col0 + 16 * f + n];
#pragma unroll
        for (int j = 0; j < 8; ++j) {
            out[(long)(row0 + 8 * g + j) * D_MODEL + col0 + 16 * f + n] = acc[f][j] + bv;
        }
    }
}

// ---------------------------------------------------------------------------
extern "C" void kernel_launch(void* const* d_in, const int* in_sizes, int n_in,
                              void* d_out, int out_size, void* d_ws, size_t ws_size,
                              hipStream_t stream) {
    const float* Q  = (const float*)d_in[0];
    const float* K  = (const float*)d_in[1];
    const float* V  = (const float*)d_in[2];
    const float* Wq = (const float*)d_in[3];
    const float* bq = (const float*)d_in[4];
    const float* Wk = (const float*)d_in[5];
    const float* bk = (const float*)d_in[6];
    const float* Wv = (const float*)d_in[7];
    const float* bv = (const float*)d_in[8];
    const float* Wo = (const float*)d_in[9];
    const float* bo = (const float*)d_in[10];
    float* out = (float*)d_out;

    // workspace carve-up (all sizes are multiples of 256B)
    char* p = (char*)d_ws;
    const size_t qkv_bytes  = (size_t)BATCH * NHEAD * SEQ * 64 * sizeof(bf16); // 8 MiB
    const size_t cat_bytes  = (size_t)BATCH * SEQ * NHEAD * DV * sizeof(bf16); // 8 MiB
    const size_t wt_bytes   = (size_t)NHEAD * 64 * D_MODEL * sizeof(bf16);     // 2 MiB
    const size_t wot_bytes  = (size_t)D_MODEL * D_MODEL * sizeof(bf16);        // 2 MiB

    bf16* qb   = (bf16*)p;  p += qkv_bytes;
    bf16* kbuf = (bf16*)p;  p += qkv_bytes;
    bf16* vTb  = (bf16*)p;  p += qkv_bytes;
    bf16* cat  = (bf16*)p;  p += cat_bytes;
    bf16* WqT  = (bf16*)p;  p += wt_bytes;
    bf16* WkT  = (bf16*)p;  p += wt_bytes;
    bf16* WvT  = (bf16*)p;  p += wt_bytes;
    bf16* WoT  = (bf16*)p;  p += wot_bytes;

    // 1) weight transposes + bf16 downconvert
    transpose_to_bf16<<<1024, 256, 0, stream>>>(Wq, WqT, NHEAD, D_MODEL, 64);
    transpose_to_bf16<<<1024, 256, 0, stream>>>(Wk, WkT, NHEAD, D_MODEL, 64);
    transpose_to_bf16<<<1024, 256, 0, stream>>>(Wv, WvT, NHEAD, D_MODEL, 64);
    transpose_to_bf16<<<1024, 256, 0, stream>>>(Wo, WoT, 1, D_MODEL, D_MODEL);

    // 2) QKV projections: 4096 waves = 512 blocks x 8 waves
    const int projBlocks = (BATCH * NHEAD * (SEQ / 16)) / 8;
    proj_kernel<0><<<projBlocks, 256, 0, stream>>>(Q, WqT, bq, qb);
    proj_kernel<0><<<projBlocks, 256, 0, stream>>>(K, WkT, bk, kbuf);
    proj_kernel<1><<<projBlocks, 256, 0, stream>>>(V, WvT, bv, vTb);

    // 3) flash attention: 4096 waves
    attn_kernel<<<projBlocks, 256, 0, stream>>>(qb, kbuf, vTb, cat);

    // 4) output projection: (4096/16 rows)*(1024/64 cols) = 4096 waves
    const int outBlocks = ((BATCH * SEQ / 16) * (D_MODEL / 64)) / 8;
    outproj_kernel<<<outBlocks, 256, 0, stream>>>(cat, WoT, bo, out);
}
